// MultiHeadAttention_17377437680318
// MI455X (gfx1250) — matrix operand order repaired and checked by reading, not compiled, not run on hardware
//
#include <hip/hip_runtime.h>
#include <hip/hip_bf16.h>

// MI455X / gfx1250: wave32, bf16 WMMA 16x16x32 with fp32 accumulation,
// async global->LDS tile copies (ASYNCcnt) when the toolchain exposes them.
typedef __bf16 bf16;
typedef __attribute__((ext_vector_type(16))) __bf16 v16bf;
typedef __attribute__((ext_vector_type(8)))  __bf16 v8bf;
typedef __attribute__((ext_vector_type(8)))  float  v8f;
typedef int gvec4i __attribute__((vector_size(16)));  // b128 payload type for async-LDS builtin

#define LN_ROPE_BASE 9.210340371976184f  /* ln(10000) */

#define WMMA_BF16(a, b, c) \
  __builtin_amdgcn_wmma_f32_16x16x32_bf16(false, (a), false, (b), (short)0, (c), false, false)

#if __has_builtin(__builtin_amdgcn_global_load_async_to_lds_b128) && \
    __has_builtin(__builtin_amdgcn_s_wait_asynccnt)
#define HAVE_ASYNC_LDS 1
#else
#define HAVE_ASYNC_LDS 0
#endif

#define AS1 __attribute__((address_space(1)))
#define AS3 __attribute__((address_space(3)))

// 16-byte global -> LDS copy; async DMA (no VGPR round trip) when available.
__device__ __forceinline__ void async_copy16(void* lds_dst, const void* gsrc) {
#if HAVE_ASYNC_LDS
  __builtin_amdgcn_global_load_async_to_lds_b128(
      (AS1 gvec4i*)gsrc, (AS3 gvec4i*)lds_dst, 0, 0);
#else
  *(v8bf*)lds_dst = *(const v8bf*)gsrc;
#endif
}
__device__ __forceinline__ void async_wait() {
#if HAVE_ASYNC_LDS
  __builtin_amdgcn_s_wait_asynccnt(0);
#endif
}

// A fragment (16x32 bf16, row-major LDS tile, leading dim ld in elements).
// Lane half h: VGPR0..3 hold K = k0+h*8 .. +7 ; VGPR4..7 hold K = k0+16+h*8 .. +7.
__device__ __forceinline__ v16bf frag_a(const bf16* base, int lane, int m0, int k0, int ld) {
  const bf16* p = base + (m0 + (lane & 15)) * ld + k0 + ((lane >> 4) << 3);
  v8bf lo = *(const v8bf*)p;
  v8bf hi = *(const v8bf*)(p + 16);
  v16bf a;
#pragma unroll
  for (int i = 0; i < 8; ++i) { a[i] = lo[i]; a[i + 8] = hi[i]; }
  return a;
}

// B fragment (32x16 bf16) read from an n-major ("Bt[n][k]") LDS tile.
// Lane half h: VGPR0..3 hold K = k0+h*8 .. +7 ; VGPR4..7 hold K = k0+16+h*8 .. +7, column n = n0 + (lane&15).
__device__ __forceinline__ v16bf frag_b(const bf16* baseT, int lane, int n0, int k0, int ld) {
  const bf16* p = baseT + (n0 + (lane & 15)) * ld + k0 + ((lane >> 4) << 3);
  v8bf lo = *(const v8bf*)p;
  v8bf hi = *(const v8bf*)(p + 16);
  v16bf b;
#pragma unroll
  for (int i = 0; i < 8; ++i) { b[i] = lo[i]; b[i + 8] = hi[i]; }
  return b;
}

// ---------------------------------------------------------------- converts
__global__ void cvt_f32_bf16_kernel(const float* __restrict__ in, bf16* __restrict__ out, int n) {
  int i = blockIdx.x * blockDim.x + threadIdx.x;
  int stride = gridDim.x * blockDim.x;
  for (; i < n; i += stride) out[i] = (bf16)in[i];
}

// ------------------------------------------- fused QKV GEMM + bias + RoPE
// X[8192,1024]bf16 @ W[1024,3072]bf16 + bias -> q/k/v bf16 in [b=4,h=16,s=2048,d=64]
__global__ __launch_bounds__(256) void qkv_rope_kernel(
    const bf16* __restrict__ X, const bf16* __restrict__ W, const float* __restrict__ bias,
    bf16* __restrict__ Qo, bf16* __restrict__ Ko, bf16* __restrict__ Vo) {
  __shared__ alignas(16) bf16 As[2][128][40];   // double-buffered 128x32 A tiles
  __shared__ alignas(16) bf16 Bt[2][128][40];   // double-buffered 32x128 B tiles, [n][k]
  __shared__ float Cs[32][132];                 // one 32x128 C chunk for RoPE epilogue

  const int tid = threadIdx.x;
  const int lane = tid & 31;
  const int wid = tid >> 5;
  const int wm = wid & 3;    // 4 waves along M (32 rows each)
  const int wn = wid >> 2;   // 2 waves along N (64 cols each)
  const int m0 = blockIdx.x * 128;
  const int n0 = blockIdx.y * 128;

  v8f zero = {};
  v8f acc[2][4];
#pragma unroll
  for (int i = 0; i < 2; ++i)
#pragma unroll
    for (int j = 0; j < 4; ++j) acc[i][j] = zero;

  const int arow = tid >> 1;        // 0..127
  const int acol = (tid & 1) << 4;  // 0,16
  const int brow = tid >> 3;        // 0..31
  const int bcol = (tid & 7) << 4;  // 0..112

  auto load_tile = [&](int kk, int buf) {
    const bf16* ga = X + (m0 + arow) * 1024 + kk + acol;
    async_copy16(&As[buf][arow][acol], ga);
    async_copy16(&As[buf][arow][acol + 8], ga + 8);
    const bf16* gb = W + (kk + brow) * 3072 + n0 + bcol;
    __builtin_prefetch(gb + 32 * 3072, 0, 1);
    v8bf x0 = *(const v8bf*)gb;
    v8bf x1 = *(const v8bf*)(gb + 8);
#pragma unroll
    for (int i = 0; i < 8; ++i) {
      Bt[buf][bcol + i][brow] = x0[i];
      Bt[buf][bcol + 8 + i][brow] = x1[i];
    }
  };

  load_tile(0, 0);
  async_wait();
  __syncthreads();

  for (int kk = 0; kk < 1024; kk += 32) {
    const int buf = (kk >> 5) & 1;
    if (kk + 32 < 1024) load_tile(kk + 32, buf ^ 1);  // prefetch next tile
#pragma unroll
    for (int ms = 0; ms < 2; ++ms) {
      v16bf a = frag_a(&As[buf][0][0], lane, wm * 32 + ms * 16, 0, 40);
#pragma unroll
      for (int ns = 0; ns < 4; ++ns) {
        v16bf b = frag_b(&Bt[buf][0][0], lane, wn * 64 + ns * 16, 0, 40);
        acc[ms][ns] = WMMA_BF16(a, b, acc[ms][ns]);
      }
    }
    async_wait();
    __syncthreads();
  }

  // Epilogue in 32-row chunks: stage to LDS so rotate_half's +/-32 partner is visible.
  const int half8 = (lane >> 4) << 3;
  for (int g = 0; g < 4; ++g) {
    __syncthreads();
    if (wm == g) {
#pragma unroll
      for (int ms = 0; ms < 2; ++ms)
#pragma unroll
        for (int ns = 0; ns < 4; ++ns)
#pragma unroll
          for (int v = 0; v < 8; ++v)
            Cs[ms * 16 + v + half8][wn * 64 + ns * 16 + (lane & 15)] = acc[ms][ns][v];
    }
    __syncthreads();
#pragma unroll 4
    for (int it = 0; it < 16; ++it) {
      int idx = it * 256 + tid;
      int r = idx >> 7, c = idx & 127;
      int mg = m0 + g * 32 + r;
      int bIdx = mg >> 11, sPos = mg & 2047;
      int f = n0 + c;
      int which = f >> 10;         // 0=q, 1=k, 2=v
      int cc = f & 1023;
      int head = cc >> 6, dd = cc & 63;
      float val = Cs[r][c] + bias[f];
      if (which < 2) {  // RoPE on q,k
        int j = dd & 31;
        float invf = __expf(-(float)(2 * j) * (1.0f / 64.0f) * LN_ROPE_BASE);
        float sn, cs;
        __sincosf((float)sPos * invf, &sn, &cs);
        int pc = (dd < 32) ? c + 32 : c - 32;
        float pv = Cs[r][pc] + bias[n0 + pc];
        float rot = (dd < 32) ? -pv : pv;
        val = val * cs + rot * sn;
      }
      long dsti = (((long)(bIdx * 16 + head)) * 2048 + sPos) * 64 + dd;
      bf16 o = (bf16)val;
      if (which == 0)      Qo[dsti] = o;
      else if (which == 1) Ko[dsti] = o;
      else                 Vo[dsti] = o;
    }
  }
}

// ----------------------------------------------------- flash attention core
// One block per (b*h, 64-row q tile). Online softmax, WMMA for QK^T and PV.
__global__ __launch_bounds__(256) void attn_kernel(
    const bf16* __restrict__ Q, const bf16* __restrict__ K, const bf16* __restrict__ V,
    bf16* __restrict__ O) {
  __shared__ alignas(16) bf16 Qs[64][72];
  __shared__ alignas(16) bf16 Ks[64][72];   // [key][dd] == B^T fragment layout for QK^T
  __shared__ alignas(16) bf16 Vt[64][72];   // [dd][key] == B^T fragment layout for PV
  __shared__ float Ss[64][68];
  __shared__ alignas(16) bf16 Ps[64][72];
  __shared__ float rowM[64], rowL[64], rowC[64];

  const int tid = threadIdx.x, lane = tid & 31, wid = tid >> 5;
  const int wm = wid >> 1;  // 4 groups of 16 rows
  const int wn = wid & 1;   // 2 halves of 32 cols
  const int qt = blockIdx.x;
  const int bh = blockIdx.y;
  const long base = (long)bh * 2048 * 64;
  const int half8 = (lane >> 4) << 3;

  const int lrow = tid >> 2;        // 0..63
  const int lcol = (tid & 3) << 4;  // 0,16,32,48

  {
    const bf16* g = Q + base + (long)(qt * 64 + lrow) * 64 + lcol;
    async_copy16(&Qs[lrow][lcol], g);
    async_copy16(&Qs[lrow][lcol + 8], g + 8);
  }
  if (tid < 64) { rowM[tid] = -3.0e38f; rowL[tid] = 0.0f; }
  v8f zero = {};
  v8f accO[2]; accO[0] = zero; accO[1] = zero;
  async_wait();
  __syncthreads();

  const float scale = 0.125f;  // 1/sqrt(64)
  for (int kt = 0; kt < 32; ++kt) {
    {
      const bf16* g = K + base + (long)(kt * 64 + lrow) * 64 + lcol;
      async_copy16(&Ks[lrow][lcol], g);
      async_copy16(&Ks[lrow][lcol + 8], g + 8);
    }
    {
      const bf16* g = V + base + (long)(kt * 64 + lrow) * 64 + lcol;
      v8bf x0 = *(const v8bf*)g;
      v8bf x1 = *(const v8bf*)(g + 8);
#pragma unroll
      for (int i = 0; i < 8; ++i) { Vt[lcol + i][lrow] = x0[i]; Vt[lcol + 8 + i][lrow] = x1[i]; }
    }
    async_wait();
    __syncthreads();

    // S = Q K^T
    v8f accS[2]; accS[0] = zero; accS[1] = zero;
#pragma unroll
    for (int kc = 0; kc < 64; kc += 32) {
      v16bf a = frag_a(&Qs[0][0], lane, wm * 16, kc, 72);
#pragma unroll
      for (int ns = 0; ns < 2; ++ns) {
        v16bf b = frag_b(&Ks[0][0], lane, wn * 32 + ns * 16, kc, 72);
        accS[ns] = WMMA_BF16(a, b, accS[ns]);
      }
    }
#pragma unroll
    for (int ns = 0; ns < 2; ++ns)
#pragma unroll
      for (int v = 0; v < 8; ++v)
        Ss[wm * 16 + v + half8][wn * 32 + ns * 16 + (lane & 15)] = accS[ns][v] * scale;
    __syncthreads();

    // online softmax (one thread per row)
    if (tid < 64) {
      int r = tid;
      float mOld = rowM[r];
      float mNew = mOld;
#pragma unroll 8
      for (int c = 0; c < 64; ++c) mNew = fmaxf(mNew, Ss[r][c]);
      float corr = __expf(mOld - mNew);
      float sum = 0.f;
#pragma unroll 8
      for (int c = 0; c < 64; ++c) {
        float p = __expf(Ss[r][c] - mNew);
        Ps[r][c] = (bf16)p;
        sum += p;
      }
      rowL[r] = rowL[r] * corr + sum;
      rowM[r] = mNew;
      rowC[r] = corr;
    }
    __syncthreads();

    // rescale running O, then O += P V
#pragma unroll
    for (int v = 0; v < 8; ++v) {
      float cf = rowC[wm * 16 + v + half8];
      accO[0][v] *= cf;
      accO[1][v] *= cf;
    }
#pragma unroll
    for (int kc = 0; kc < 64; kc += 32) {
      v16bf a = frag_a(&Ps[0][0], lane, wm * 16, kc, 72);
#pragma unroll
      for (int ns = 0; ns < 2; ++ns) {
        v16bf b = frag_b(&Vt[0][0], lane, wn * 32 + ns * 16, kc, 72);
        accO[ns] = WMMA_BF16(a, b, accO[ns]);
      }
    }
    __syncthreads();
  }

  // normalize and write [b, s, h*64+dd] bf16
  const int bIdx = bh >> 4, h = bh & 15;
#pragma unroll
  for (int ns = 0; ns < 2; ++ns)
#pragma unroll
    for (int v = 0; v < 8; ++v) {
      int r = wm * 16 + v + half8;
      int sPos = qt * 64 + r;
      int dd = wn * 32 + ns * 16 + (lane & 15);
      float val = accO[ns][v] / rowL[r];
      O[((long)(bIdx * 2048 + sPos)) * 1024 + h * 64 + dd] = (bf16)val;
    }
}

// ------------------------------------------------ output projection GEMM
// A[8192,1024]bf16 @ Wo[1024,1024]bf16 + bo -> fp32 out
__global__ __launch_bounds__(256) void out_proj_kernel(
    const bf16* __restrict__ A, const bf16* __restrict__ W, const float* __restrict__ bias,
    float* __restrict__ Out) {
  __shared__ alignas(16) bf16 As[2][128][40];
  __shared__ alignas(16) bf16 Bt[2][128][40];
  const int tid = threadIdx.x, lane = tid & 31, wid = tid >> 5;
  const int wm = wid & 3, wn = wid >> 2;
  const int m0 = blockIdx.x * 128, n0 = blockIdx.y * 128;
  const int half8 = (lane >> 4) << 3;

  v8f zero = {};
  v8f acc[2][4];
#pragma unroll
  for (int i = 0; i < 2; ++i)
#pragma unroll
    for (int j = 0; j < 4; ++j) acc[i][j] = zero;

  const int arow = tid >> 1;
  const int acol = (tid & 1) << 4;
  const int brow = tid >> 3;
  const int bcol = (tid & 7) << 4;

  auto load_tile = [&](int kk, int buf) {
    const bf16* ga = A + (long)(m0 + arow) * 1024 + kk + acol;
    async_copy16(&As[buf][arow][acol], ga);
    async_copy16(&As[buf][arow][acol + 8], ga + 8);
    const bf16* gb = W + (kk + brow) * 1024 + n0 + bcol;
    __builtin_prefetch(gb + 32 * 1024, 0, 1);
    v8bf x0 = *(const v8bf*)gb;
    v8bf x1 = *(const v8bf*)(gb + 8);
#pragma unroll
    for (int i = 0; i < 8; ++i) {
      Bt[buf][bcol + i][brow] = x0[i];
      Bt[buf][bcol + 8 + i][brow] = x1[i];
    }
  };

  load_tile(0, 0);
  async_wait();
  __syncthreads();

  for (int kk = 0; kk < 1024; kk += 32) {
    const int buf = (kk >> 5) & 1;
    if (kk + 32 < 1024) load_tile(kk + 32, buf ^ 1);
#pragma unroll
    for (int ms = 0; ms < 2; ++ms) {
      v16bf a = frag_a(&As[buf][0][0], lane, wm * 32 + ms * 16, 0, 40);
#pragma unroll
      for (int ns = 0; ns < 4; ++ns) {
        v16bf b = frag_b(&Bt[buf][0][0], lane, wn * 64 + ns * 16, 0, 40);
        acc[ms][ns] = WMMA_BF16(a, b, acc[ms][ns]);
      }
    }
    async_wait();
    __syncthreads();
  }

#pragma unroll
  for (int ms = 0; ms < 2; ++ms)
#pragma unroll
    for (int ns = 0; ns < 4; ++ns)
#pragma unroll
      for (int v = 0; v < 8; ++v) {
        int mg = m0 + wm * 32 + ms * 16 + v + half8;
        int ng = n0 + wn * 64 + ns * 16 + (lane & 15);
        Out[(long)mg * 1024 + ng] = acc[ms][ns][v] + bias[ng];
      }
}

// ---------------------------------------------------------------- launcher
extern "C" void kernel_launch(void* const* d_in, const int* in_sizes, int n_in,
                              void* d_out, int out_size, void* d_ws, size_t ws_size,
                              hipStream_t stream) {
  const float* x    = (const float*)d_in[0];  // [4,2048,1024]
  const float* Wqkv = (const float*)d_in[1];  // [1024,3072]
  const float* bqkv = (const float*)d_in[2];  // [3072]
  const float* Wo   = (const float*)d_in[3];  // [1024,1024]
  const float* bo   = (const float*)d_in[4];  // [1024]
  float* out = (float*)d_out;                 // [4,2048,1024]

  // Workspace layout (~88 MB total)
  char* ws = (char*)d_ws;
  size_t off = 0;
  bf16* xb  = (bf16*)(ws + off); off += (size_t)8192 * 1024 * 2;      // x bf16
  bf16* wqb = (bf16*)(ws + off); off += (size_t)1024 * 3072 * 2;      // Wqkv bf16
  bf16* wob = (bf16*)(ws + off); off += (size_t)1024 * 1024 * 2;      // Wo bf16
  bf16* qb  = (bf16*)(ws + off); off += (size_t)64 * 2048 * 64 * 2;   // q [b,h,s,d]
  bf16* kb  = (bf16*)(ws + off); off += (size_t)64 * 2048 * 64 * 2;   // k
  bf16* vb  = (bf16*)(ws + off); off += (size_t)64 * 2048 * 64 * 2;   // v
  bf16* ao  = (bf16*)(ws + off); off += (size_t)8192 * 1024 * 2;      // attn out [b,s,e]

  cvt_f32_bf16_kernel<<<4096, 256, 0, stream>>>(x,    xb,  8192 * 1024);
  cvt_f32_bf16_kernel<<<4096, 256, 0, stream>>>(Wqkv, wqb, 1024 * 3072);
  cvt_f32_bf16_kernel<<<2048, 256, 0, stream>>>(Wo,   wob, 1024 * 1024);

  qkv_rope_kernel<<<dim3(64, 24), 256, 0, stream>>>(xb, wqb, bqkv, qb, kb, vb);
  attn_kernel<<<dim3(32, 64), 256, 0, stream>>>(qb, kb, vb, ao);
  out_proj_kernel<<<dim3(64, 8), 256, 0, stream>>>(ao, wob, bo, out);
}
